// vox_fea_46480136077752
// MI455X (gfx1250) — compile-verified, matrix operand order checked by reading, not run on hardware
//
#include <hip/hip_runtime.h>
#include <hip/hip_bf16.h>

// MI455X voxel-feature extractor.
// Memory-bound (~830MB moved vs ~46 GFLOP at f16): f16 activations,
// BN-affine+ReLU fused into GEMM A-tile staging (vectorized b128 loads),
// BN batch-stats fused into GEMM epilogue (wave->LDS->global reduction),
// scatter-mean fused into the last GEMM. Matrix math is
// v_wmma_f32_16x16x32_f16 with B pre-packed into the ISA per-lane fragment
// layout so every B fragment is one contiguous 32B load (L2/WGP$ resident).
// All hot-path addressing uses 32-bit offsets so stores lower to
// saddr + voffset form instead of per-lane 64-bit adds.

typedef __attribute__((ext_vector_type(16))) _Float16 v16h;
typedef __attribute__((ext_vector_type(8)))  _Float16 v8h;
typedef __attribute__((ext_vector_type(8)))  float    v8f;

#define VF_GX   480
#define VF_GY   360
#define VF_GXY  (VF_GX * VF_GY)
#define VF_B    2
#define VF_N    200000
#define VF_P    (VF_B * VF_N)      // 400000, divisible by 64
#define VF_NKEYS (VF_B * VF_GXY)   // 345600 < P
#define VF_EPS  1e-5f

__device__ __forceinline__ v8f vf_wmma(v16h a, v16h b, v8f c) {
  return __builtin_amdgcn_wmma_f32_16x16x32_f16(false, a, false, b,
                                                (short)0, c, false, false);
}

// ---------------------------------------------------------------- init ----
__global__ void vf_init(float* stats, int* flags, float* counts,
                        float* pooled, long long* unq) {
  const long long total = (long long)VF_P * 64;
  for (long long i = blockIdx.x * (long long)blockDim.x + threadIdx.x;
       i < total; i += (long long)gridDim.x * blockDim.x) {
    pooled[i] = 0.0f;
    if (i < 2048)      stats[i] = 0.0f;
    if (i < VF_NKEYS)  flags[i] = 0;
    if (i < VF_P) {
      counts[i] = 0.0f;
      // padded jnp.unique rows: key=-1 -> floordiv/mod decode = (-1,479,359)
      unq[3 * i + 0] = -1;
      unq[3 * i + 1] = VF_GX - 1;
      unq[3 * i + 2] = VF_GY - 1;
    }
  }
}

// ------------------------------------------------- weight fragment pack ---
// Pack W (Kin x N, row-major fp32) into per-lane B-fragment order for
// v_wmma_f32_16x16x32_f16: for tile (ntg,kt), lane L holds column n=ntg*16+
// (L&15); K entries { hi*8 + j , 16 + hi*8 + j } (hi = L>>4). Rows >= Kin pad 0.
__global__ void vf_pack_w(const float* __restrict__ W, _Float16* __restrict__ out,
                          int Kin, int KP, int N) {
  int KT = KP >> 5, NT = N >> 4;
  int t = blockIdx.x * blockDim.x + threadIdx.x;
  int total = NT * KT * 32;
  if (t >= total) return;
  int lane = t & 31, tile = t >> 5;
  int kt = tile % KT, ntg = tile / KT;
  int lr = lane & 15, hi = (lane >> 4) & 1;
  int n = ntg * 16 + lr;
  #pragma unroll
  for (int j = 0; j < 16; ++j) {
    int k = kt * 32 + hi * 8 + j + ((j >= 8) ? 8 : 0);
    float v = (k < Kin) ? W[(unsigned)(k * N + n)] : 0.0f;
    out[(unsigned)((tile * 32 + lane) * 16 + j)] = (_Float16)v;
  }
}

// --------------------------------------------------------- voxel keys ----
__global__ void vf_keys(const long long* __restrict__ xy, int* __restrict__ keys,
                        int* __restrict__ flags) {
  int p = blockIdx.x * blockDim.x + threadIdx.x;
  if (p >= VF_P) return;
  int b = p / VF_N;
  int ix = (int)xy[2 * (size_t)p + 0];
  int iy = (int)xy[2 * (size_t)p + 1];
  int key = b * VF_GXY + ix * VF_GY + iy;
  keys[p] = key;
  flags[key] = 1;              // benign race, all writers store 1
}

// ---------------------------------------- prefix scan over dense flags ----
__global__ void vf_scan1(const int* __restrict__ flags, int* __restrict__ pos,
                         int* __restrict__ bsums) {
  __shared__ int sh[256];
  int tid = threadIdx.x;
  int base = blockIdx.x * 1024 + tid * 4;
  int v[4]; int s = 0;
  #pragma unroll
  for (int i = 0; i < 4; ++i) {
    v[i] = (base + i < VF_NKEYS) ? flags[base + i] : 0;
    s += v[i];
  }
  sh[tid] = s; __syncthreads();
  for (int off = 1; off < 256; off <<= 1) {
    int t = (tid >= off) ? sh[tid - off] : 0;
    __syncthreads();
    sh[tid] += t;
    __syncthreads();
  }
  if (tid == 255) bsums[blockIdx.x] = sh[255];
  int run = sh[tid] - s;       // exclusive within block
  #pragma unroll
  for (int i = 0; i < 4; ++i) {
    if (base + i < VF_NKEYS) pos[base + i] = run;
    run += v[i];
  }
}

__global__ void vf_scan2(int* __restrict__ bsums, int nb) {
  __shared__ int sh[512];
  int tid = threadIdx.x;
  int v = (tid < nb) ? bsums[tid] : 0;
  sh[tid] = v; __syncthreads();
  for (int off = 1; off < 512; off <<= 1) {
    int t = (tid >= off) ? sh[tid - off] : 0;
    __syncthreads();
    sh[tid] += t;
    __syncthreads();
  }
  if (tid < nb) bsums[tid] = sh[tid] - v;   // exclusive
}

__global__ void vf_scan3(int* __restrict__ pos, const int* __restrict__ flags,
                         const int* __restrict__ bsums, long long* __restrict__ unq) {
  int i = blockIdx.x * blockDim.x + threadIdx.x;
  if (i >= VF_NKEYS) return;
  int p = pos[i] + bsums[i >> 10];
  pos[i] = p;
  if (flags[i]) {
    int b = i / VF_GXY, rem = i % VF_GXY;
    unq[3 * (size_t)p + 0] = b;
    unq[3 * (size_t)p + 1] = rem / VF_GY;
    unq[3 * (size_t)p + 2] = rem % VF_GY;
  }
}

__global__ void vf_inv(const int* __restrict__ keys, const int* __restrict__ pos,
                       int* __restrict__ inv, float* __restrict__ counts) {
  int p = blockIdx.x * blockDim.x + threadIdx.x;
  if (p >= VF_P) return;
  int u = pos[keys[p]];
  inv[p] = u;
  atomicAdd(&counts[u], 1.0f);
}

// ------------------------------------------------------------ BN stats ----
__global__ void vf_bn0_stats(const float* __restrict__ x,
                             float* __restrict__ gsum, float* __restrict__ gsq) {
  __shared__ float ss[9], sq[9];
  int tid = threadIdx.x;
  if (tid < 9) { ss[tid] = 0.0f; sq[tid] = 0.0f; }
  __syncthreads();
  float ls[9], lq[9];
  #pragma unroll
  for (int f = 0; f < 9; ++f) { ls[f] = 0.0f; lq[f] = 0.0f; }
  for (int p = blockIdx.x * blockDim.x + tid; p < VF_P;
       p += gridDim.x * blockDim.x) {
    #pragma unroll
    for (int f = 0; f < 9; ++f) {
      float v = x[(unsigned)(p * 9 + f)];
      ls[f] += v; lq[f] += v * v;
    }
  }
  #pragma unroll
  for (int f = 0; f < 9; ++f) { atomicAdd(&ss[f], ls[f]); atomicAdd(&sq[f], lq[f]); }
  __syncthreads();
  if (tid < 9) { atomicAdd(&gsum[tid], ss[tid]); atomicAdd(&gsq[tid], sq[tid]); }
}

// a = g * rsqrt(var+eps);  c = b - mean * a   (train-mode, biased variance)
__global__ void vf_bn_finalize(const float* __restrict__ sum, const float* __restrict__ sq,
                               const float* __restrict__ g, const float* __restrict__ b,
                               float* __restrict__ a, float* __restrict__ c, int F) {
  int i = blockIdx.x * blockDim.x + threadIdx.x;
  if (i >= F) return;
  const float invP = 1.0f / (float)VF_P;
  float m = sum[i] * invP;
  float var = sq[i] * invP - m * m;
  float r = rsqrtf(var + VF_EPS);
  float ai = g[i] * r;
  a[i] = ai;
  c[i] = b[i] - m * ai;
}

// --------------------------------------------------------- fused GEMM -----
// Block: 64 rows x NOUT cols, 8 waves = (4 m-tiles) x (2 n-halves).
// MI: 0 = fp32 input, BN-affine (layer1); 1 = f16 input, BN-affine + ReLU.
// MO: 0 = write f16 h + per-feature sum/sum^2 (wave->LDS->global);
//     1 = bias + scatter-add into pooled output.
// P % 64 == 0, so no row bounds checks anywhere (straight-line epilogue).
template <int KP, int NOUT, int MI, int MO>
__global__ __launch_bounds__(256) void vf_gemm(
    const void* __restrict__ inA, int Kin, int inStride,
    const _Float16* __restrict__ Wpack, const float* __restrict__ bias,
    const float* __restrict__ affA, const float* __restrict__ affC,
    _Float16* __restrict__ hout, float* __restrict__ gsum, float* __restrict__ gsq,
    const int* __restrict__ inv, float* __restrict__ pooled) {
  constexpr int LDK  = KP + 16;           // A row stride (f16), 32B multiple
  constexpr int NT   = NOUT / 16;
  constexpr int NTW  = NT / 2;            // n-tiles per wave
  constexpr int KT   = KP / 32;
  constexpr int REDN = (MO == 0) ? NOUT : 1;
  __shared__ __align__(32) _Float16 As[64 * LDK];
  __shared__ float sAf[KP], sCf[KP];
  __shared__ float redS[REDN], redQ[REDN];

  const int tid  = threadIdx.x;
  const int lane = tid & 31;
  const int w    = tid >> 5;
  const int mt   = w & 3;
  const int nh   = w >> 2;
  const unsigned row0 = blockIdx.x * 64u;

  __builtin_prefetch(Wpack + (unsigned)lane * 64u, 0, 3); // global_prefetch_b8

  // Stage BN affine coefficients (zero-padded) and zero stat reduction.
  for (int k = tid; k < KP; k += 256) {
    sAf[k] = (k < Kin) ? affA[k] : 0.0f;
    sCf[k] = (k < Kin) ? affC[k] : 0.0f;
  }
  if (MO == 0)
    for (int k = tid; k < NOUT; k += 256) { redS[k] = 0.0f; redQ[k] = 0.0f; }
  __syncthreads();

  // Stage A tile: BN affine (+ReLU for hidden layers), cast to f16.
  if (MI == 0) {
    for (int idx = tid; idx < 64 * KP; idx += 256) {
      int r = idx / KP, k = idx - r * KP;
      float v = 0.0f;
      if (k < Kin) {
        float x = ((const float*)inA)[(row0 + r) * (unsigned)inStride + k];
        v = x * sAf[k] + sCf[k];
      }
      As[r * LDK + k] = (_Float16)v;
    }
  } else {
    constexpr int C8 = KP / 8;            // 16B chunks per row (Kin == KP)
    for (int idx = tid; idx < 64 * C8; idx += 256) {
      int r = idx / C8, k0 = (idx - r * C8) * 8;
      v8h x = *(const v8h*)((const _Float16*)inA +
                            (row0 + r) * (unsigned)inStride + k0);
      v8h o;
      #pragma unroll
      for (int j = 0; j < 8; ++j) {
        float v = (float)x[j] * sAf[k0 + j] + sCf[k0 + j];
        o[j] = (_Float16)(v > 0.0f ? v : 0.0f);
      }
      *(v8h*)&As[r * LDK + k0] = o;
    }
  }
  __syncthreads();

  v8f acc[NTW];
  #pragma unroll
  for (int t = 0; t < NTW; ++t)
    #pragma unroll
    for (int i = 0; i < 8; ++i) acc[t][i] = 0.0f;

  const int lr = lane & 15, hi = (lane >> 4) & 1;
  #pragma unroll
  for (int kt = 0; kt < KT; ++kt) {
    // A fragment per ISA layout: lane<16 -> K {0..7,16..23}, lane>=16 -> {8..15,24..31}
    const _Float16* ap = &As[(mt * 16 + lr) * LDK + kt * 32 + hi * 8];
    v8h alo = *(const v8h*)ap;
    v8h ahi = *(const v8h*)(ap + 16);
    v16h a;
    #pragma unroll
    for (int i = 0; i < 8; ++i) { a[i] = alo[i]; a[i + 8] = ahi[i]; }
    #pragma unroll
    for (int nt = 0; nt < NTW; ++nt) {
      int ntg = nh * NTW + nt;
      v16h b = *(const v16h*)(Wpack + (unsigned)((ntg * KT + kt) * 32 + lane) * 16u);
      acc[nt] = vf_wmma(a, b, acc[nt]);
    }
  }

  // Epilogue. C/D layout: VGPR r -> row mt*16 + r + 8*hi, col = (nh*NTW+nt)*16 + lr.
  #pragma unroll
  for (int nt = 0; nt < NTW; ++nt) {
    int n = (nh * NTW + nt) * 16 + lr;
    float bn = bias[n];
    float s = 0.0f, q = 0.0f;
    const unsigned rbase = row0 + (unsigned)(mt * 16 + 8 * hi);
    #pragma unroll
    for (int r = 0; r < 8; ++r) {
      unsigned grow = rbase + (unsigned)r;
      float v = acc[nt][r] + bn;
      if (MO == 0) {
        hout[grow * (unsigned)NOUT + (unsigned)n] = (_Float16)v;
        s += v; q += v * v;
      } else {
        unsigned u = (unsigned)inv[grow];
        atomicAdd(&pooled[u * (unsigned)NOUT + (unsigned)n], v);
      }
    }
    if (MO == 0) {
      s += __shfl_xor(s, 16, 32);   // combine hi/lo half-rows (wave32)
      q += __shfl_xor(q, 16, 32);
      if (hi == 0) { atomicAdd(&redS[n], s); atomicAdd(&redQ[n], q); }
    }
  }
  if (MO == 0) {
    __syncthreads();
    for (int n = tid; n < NOUT; n += 256) {
      atomicAdd(&gsum[n], redS[n]);
      atomicAdd(&gsq[n], redQ[n]);
    }
  }
}

// --------------------------------------------------------- scatter mean ---
__global__ void vf_divide(float* __restrict__ pooled, const float* __restrict__ counts) {
  const unsigned total = (unsigned)VF_P * 64u;
  for (unsigned i = blockIdx.x * blockDim.x + threadIdx.x;
       i < total; i += gridDim.x * blockDim.x) {
    float c = counts[i >> 6];
    pooled[i] /= fmaxf(c, 1.0f);
  }
}

// -------------------------------------------------------------- launch ----
static inline size_t vf_align(size_t x) { return (x + 255) & ~(size_t)255; }

extern "C" void kernel_launch(void* const* d_in, const int* in_sizes, int n_in,
                              void* d_out, int out_size, void* d_ws, size_t ws_size,
                              hipStream_t stream) {
  (void)in_sizes; (void)n_in; (void)out_size; (void)ws_size;
  const float*     pt   = (const float*)d_in[0];
  const long long* xy   = (const long long*)d_in[1];   // int64
  const float* bn0g = (const float*)d_in[2],  *bn0b = (const float*)d_in[3];
  const float* W1 = (const float*)d_in[4],  *b1 = (const float*)d_in[5];
  const float* bn1g = (const float*)d_in[6],  *bn1b = (const float*)d_in[7];
  const float* W2 = (const float*)d_in[8],  *b2 = (const float*)d_in[9];
  const float* bn2g = (const float*)d_in[10], *bn2b = (const float*)d_in[11];
  const float* W3 = (const float*)d_in[12], *b3 = (const float*)d_in[13];
  const float* bn3g = (const float*)d_in[14], *bn3b = (const float*)d_in[15];
  const float* W4 = (const float*)d_in[16], *b4 = (const float*)d_in[17];

  long long* unq   = (long long*)d_out;                      // [P,3] int64
  float* pooled    = (float*)((char*)d_out + (size_t)VF_P * 3 * sizeof(long long));

  // workspace layout
  char* ws = (char*)d_ws;
  float* stats = (float*)ws;                 // 2048 floats (sums + affines)
  float *sum0 = stats,        *sq0 = stats + 16;
  float *sum1 = stats + 32,   *sq1 = stats + 96;
  float *sum2 = stats + 160,  *sq2 = stats + 288;
  float *sum3 = stats + 416,  *sq3 = stats + 672;
  float *a0 = stats + 1024, *c0 = a0 + 16;
  float *a1 = c0 + 16,  *c1 = a1 + 64;
  float *a2 = c1 + 64,  *c2 = a2 + 128;
  float *a3 = c2 + 128, *c3 = a3 + 256;
  size_t off = vf_align(2048 * sizeof(float));
  _Float16* w1p = (_Float16*)(ws + off); off = vf_align(off + (size_t)(64/16)*(32/32)*32*16 * 2);
  _Float16* w2p = (_Float16*)(ws + off); off = vf_align(off + (size_t)(128/16)*(64/32)*32*16 * 2);
  _Float16* w3p = (_Float16*)(ws + off); off = vf_align(off + (size_t)(256/16)*(128/32)*32*16 * 2);
  _Float16* w4p = (_Float16*)(ws + off); off = vf_align(off + (size_t)(64/16)*(256/32)*32*16 * 2);
  int*   keys   = (int*)(ws + off);   off = vf_align(off + (size_t)VF_P * 4);
  int*   inv    = (int*)(ws + off);   off = vf_align(off + (size_t)VF_P * 4);
  float* counts = (float*)(ws + off); off = vf_align(off + (size_t)VF_P * 4);
  int*   flags  = (int*)(ws + off);   off = vf_align(off + (size_t)VF_NKEYS * 4);
  int*   pos    = (int*)(ws + off);   off = vf_align(off + (size_t)VF_NKEYS * 4);
  int*   bsums  = (int*)(ws + off);   off = vf_align(off + 512 * 4);
  _Float16* hA  = (_Float16*)(ws + off); off = vf_align(off + (size_t)VF_P * 256 * 2);
  _Float16* hB  = (_Float16*)(ws + off); off = vf_align(off + (size_t)VF_P * 128 * 2);

  const int NB = (VF_NKEYS + 1023) / 1024;        // 338 scan blocks
  const int GP = (VF_P + 255) / 256;
  const int GB = VF_P / 64;                       // 6250 GEMM blocks

  vf_init<<<2048, 256, 0, stream>>>(stats, flags, counts, pooled, unq);

  vf_pack_w<<<1, 256, 0, stream>>>(W1, w1p, 9, 32, 64);
  vf_pack_w<<<2, 256, 0, stream>>>(W2, w2p, 64, 64, 128);
  vf_pack_w<<<8, 256, 0, stream>>>(W3, w3p, 128, 128, 256);
  vf_pack_w<<<4, 256, 0, stream>>>(W4, w4p, 256, 256, 64);

  vf_keys<<<GP, 256, 0, stream>>>(xy, keys, flags);
  vf_scan1<<<NB, 256, 0, stream>>>(flags, pos, bsums);
  vf_scan2<<<1, 512, 0, stream>>>(bsums, NB);
  vf_scan3<<<(VF_NKEYS + 255) / 256, 256, 0, stream>>>(pos, flags, bsums, unq);
  vf_inv<<<GP, 256, 0, stream>>>(keys, pos, inv, counts);

  vf_bn0_stats<<<1024, 256, 0, stream>>>(pt, sum0, sq0);
  vf_bn_finalize<<<1, 256, 0, stream>>>(sum0, sq0, bn0g, bn0b, a0, c0, 9);

  vf_gemm<32, 64, 0, 0><<<GB, 256, 0, stream>>>(pt, 9, 9, w1p, b1, a0, c0,
                                                hA, sum1, sq1, nullptr, nullptr);
  vf_bn_finalize<<<1, 256, 0, stream>>>(sum1, sq1, bn1g, bn1b, a1, c1, 64);

  vf_gemm<64, 128, 1, 0><<<GB, 256, 0, stream>>>(hA, 64, 64, w2p, b2, a1, c1,
                                                 hB, sum2, sq2, nullptr, nullptr);
  vf_bn_finalize<<<1, 256, 0, stream>>>(sum2, sq2, bn2g, bn2b, a2, c2, 128);

  vf_gemm<128, 256, 1, 0><<<GB, 256, 0, stream>>>(hB, 128, 128, w3p, b3, a2, c2,
                                                  hA, sum3, sq3, nullptr, nullptr);
  vf_bn_finalize<<<1, 256, 0, stream>>>(sum3, sq3, bn3g, bn3b, a3, c3, 256);

  vf_gemm<256, 64, 1, 1><<<GB, 256, 0, stream>>>(hA, 256, 256, w4p, b4, a3, c3,
                                                 nullptr, nullptr, nullptr, inv, pooled);

  vf_divide<<<2048, 256, 0, stream>>>(pooled, counts);
}
